// GPTAttention_67851893342614
// MI455X (gfx1250) — compile-verified, matrix operand order checked by reading, not run on hardware
//
#include <hip/hip_runtime.h>

// ---------------------------------------------------------------------------
// GPT attention block for MI455X (gfx1250, wave32, WMMA bf16 + TDM path)
//   B=8, T=1024, E=768, H=12, D=64
// ---------------------------------------------------------------------------

#define Bc 8
#define Tc 1024
#define Ec 768
#define Hc 12
#define Dc 64
#define E3 (3 * Ec)     // 2304
#define Mrows (Bc * Tc) // 8192

typedef __attribute__((ext_vector_type(16))) __bf16       v16bf;
typedef __attribute__((ext_vector_type(8)))  float        v8f;
typedef __attribute__((ext_vector_type(4)))  unsigned int u32x4;
typedef __attribute__((ext_vector_type(8)))  int          i32x8;
typedef __attribute__((ext_vector_type(4)))  int          i32x4;

union FragBF {
    v16bf          v;
    unsigned short u[16];
    u32x4          q4[2];
};

// Native bf16 convert (RTNE) — let the backend pick v_cvt_*_bf16_f32
__device__ __forceinline__ unsigned short f2bf(float f) {
    __bf16 h = (__bf16)f;
    return __builtin_bit_cast(unsigned short, h);
}

__device__ __forceinline__ v8f wmma_bf16(const FragBF& a, const FragBF& b, v8f c) {
    return __builtin_amdgcn_wmma_f32_16x16x32_bf16(
        /*neg_a=*/false, a.v, /*neg_b=*/false, b.v,
        /*c_mod=*/(short)0, c, /*reuse_a=*/false, /*reuse_b=*/false);
}

#define NEG_BIG (-3.0e38f)

// ---------------------------------------------------------------------------
// Tensor Data Mover helpers (ISA cdna5 §8: D# descriptor, groups 0/1; 2-D tile)
// Toolchain uses the 6-arg builtin: (g0 u32x4, g1 i32x8, g2 i32x4, g3 i32x4,
//                                    gX i32x8, cpol i32)
// ---------------------------------------------------------------------------
typedef const __attribute__((address_space(3))) void* lds_cptr_t;

__device__ __forceinline__ unsigned lds_off(const void* p) {
    return (unsigned)(unsigned long long)(lds_cptr_t)p;   // addrspacecast -> LDS byte offset
}

// 2-D tile load: tile_d0 contiguous elements per row, tile_d1 rows,
// row stride = stride0 (data_size units). data_size = 2 bytes (bf16).
// LDS padding per ISA: interval 2^(pi+1) DWORDs, amount (pa+1) DWORDs.
__device__ __forceinline__ void tdm_load_2d(unsigned lds_byte, const void* gp,
                                            unsigned tensor_d0, unsigned tensor_d1,
                                            unsigned tile_d0, unsigned tile_d1,
                                            unsigned stride0,
                                            unsigned pad_int_code, unsigned pad_amt_code) {
    unsigned long long ga = (unsigned long long)gp;
    u32x4 g0;
    g0[0] = 1u;                                            // count=1 (valid), user mode
    g0[1] = lds_byte;                                      // lds_addr
    g0[2] = (unsigned)ga;                                  // global_addr[31:0]
    g0[3] = (unsigned)((ga >> 32) & 0x1FFFFFFu) | (2u << 30); // global_addr[56:32], type=2
    i32x8 g1;
    g1[0] = (int)((1u << 16)                               // data_size = 1 -> 2 bytes
                  | (1u << 20)                             // pad_enable
                  | (pad_int_code << 22)
                  | (pad_amt_code << 25));
    g1[1] = (int)((tensor_d0 & 0xFFFFu) << 16);            // tensor_dim0[15:0] (barrier addr 0)
    g1[2] = (int)(((tensor_d0 >> 16) & 0xFFFFu) | ((tensor_d1 & 0xFFFFu) << 16));
    g1[3] = (int)(((tensor_d1 >> 16) & 0xFFFFu) | ((tile_d0 & 0xFFFFu) << 16));
    g1[4] = (int)(tile_d1 & 0xFFFFu);                      // tile_dim1 (tile_dim2 = 0)
    g1[5] = (int)stride0;                                  // tensor_dim0_stride[31:0]
    g1[6] = 0;
    g1[7] = 0;
    i32x4 z4 = {0, 0, 0, 0};
    i32x8 z8 = {0, 0, 0, 0, 0, 0, 0, 0};
    __builtin_amdgcn_tensor_load_to_lds(g0, g1, z4, z4, z8, 0);
}

// ---------------------------------------------------------------------------
// Kernel 0: transpose + downcast weights to bf16, Wt[N][K] layout
// ---------------------------------------------------------------------------
__global__ void convert_weights(const float* __restrict__ w_attn,
                                const float* __restrict__ w_proj,
                                unsigned short* __restrict__ wt_attn,
                                unsigned short* __restrict__ wt_proj) {
    const int NA = E3 * Ec;
    const int NP = Ec * Ec;
    int idx = blockIdx.x * 256 + threadIdx.x;
    if (idx < NA) {
        int n = idx / Ec, k = idx % Ec;
        wt_attn[idx] = f2bf(w_attn[(size_t)k * E3 + n]);
    } else if (idx < NA + NP) {
        int t = idx - NA;
        int n = t / Ec, k = t % Ec;
        wt_proj[t] = f2bf(w_proj[(size_t)k * Ec + n]);
    }
}

// ---------------------------------------------------------------------------
// Kernel 1: QKV GEMM  qkv = x @ w_attn + b_attn, scattered into
//   Qh[B,H,T,D] bf16, Kh[B,H,T,D] bf16, Vt[B,H,D,T] bf16 (V transposed)
// Grid: (8192/32, 2304/256), 256 thr (8 waves, 2Mx4N), wave tile 16x64
// ---------------------------------------------------------------------------
__global__ void qkv_gemm(const float* __restrict__ x,
                         const unsigned short* __restrict__ wt,
                         const float* __restrict__ bias,
                         unsigned short* __restrict__ Qh,
                         unsigned short* __restrict__ Kh,
                         unsigned short* __restrict__ Vt) {
    const int lane = threadIdx.x & 31;
    const int wave = threadIdx.x >> 5;
    const int m0 = blockIdx.x * 32 + (wave >> 2) * 16;
    const int n0 = blockIdx.y * 256 + (wave & 3) * 64;
    const int lm = lane & 15;
    const int hi = lane >> 4;
    const int kb = hi * 8;

    v8f acc0 = {}, acc1 = {}, acc2 = {}, acc3 = {};
    const float* arow = x + (size_t)(m0 + lm) * Ec;

    for (int kk = 0; kk < Ec; kk += 32) {
        FragBF a;
#pragma unroll
        for (int j = 0; j < 8; ++j) {
            a.u[j]     = f2bf(arow[kk + kb + j]);
            a.u[8 + j] = f2bf(arow[kk + kb + 16 + j]);
        }
        FragBF b;
        const unsigned short* w0 = wt + (size_t)(n0 + lm) * Ec + kk;
        b.q4[0] = *(const u32x4*)(w0 + kb);
        b.q4[1] = *(const u32x4*)(w0 + kb + 16);
        acc0 = wmma_bf16(a, b, acc0);
        const unsigned short* w1 = wt + (size_t)(n0 + 16 + lm) * Ec + kk;
        b.q4[0] = *(const u32x4*)(w1 + kb);
        b.q4[1] = *(const u32x4*)(w1 + kb + 16);
        acc1 = wmma_bf16(a, b, acc1);
        const unsigned short* w2 = wt + (size_t)(n0 + 32 + lm) * Ec + kk;
        b.q4[0] = *(const u32x4*)(w2 + kb);
        b.q4[1] = *(const u32x4*)(w2 + kb + 16);
        acc2 = wmma_bf16(a, b, acc2);
        const unsigned short* w3 = wt + (size_t)(n0 + 48 + lm) * Ec + kk;
        b.q4[0] = *(const u32x4*)(w3 + kb);
        b.q4[1] = *(const u32x4*)(w3 + kb + 16);
        acc3 = wmma_bf16(a, b, acc3);
    }

#pragma unroll
    for (int nt = 0; nt < 4; ++nt) {
        v8f acc = (nt == 0) ? acc0 : (nt == 1) ? acc1 : (nt == 2) ? acc2 : acc3;
        int col = n0 + nt * 16 + lm;
        float bv = bias[col];
        int sec  = col / Ec;       // 0=q 1=k 2=v
        int cc   = col % Ec;
        int head = cc >> 6;
        int dd   = cc & 63;
#pragma unroll
        for (int r = 0; r < 8; ++r) {
            int row  = m0 + hi * 8 + r;
            int bidx = row >> 10;
            int tt   = row & (Tc - 1);
            unsigned short h = f2bf(acc[r] + bv);
            if (sec == 2) {
                Vt[((size_t)(bidx * Hc + head) * Dc + dd) * Tc + tt] = h;
            } else {
                size_t idx = ((size_t)(bidx * Hc + head) * Tc + tt) * Dc + dd;
                if (sec == 0) Qh[idx] = h; else Kh[idx] = h;
            }
        }
    }
}

// ---------------------------------------------------------------------------
// Kernel 2: causal flash attention with TDM double-buffered K/V staging.
// Block: 256 thr = 8 waves; wave w owns q rows [qb0+16w, +16).
// Wave 0 issues tensor_load_to_lds for chunk i+1 while all compute chunk i.
// Grid: (T/128, B*H) = (8, 96)
// ---------------------------------------------------------------------------
__global__ void attn_kernel(const unsigned short* __restrict__ Qh,
                            const unsigned short* __restrict__ Kh,
                            const unsigned short* __restrict__ Vt,
                            unsigned short* __restrict__ attn_out) {
    __shared__ __align__(16) unsigned short Klds[2][32][72];   // 32 keys x 64 d (+4dw pad/row)
    __shared__ __align__(16) unsigned short Vlds[2][64][40];   // 64 d x 32 keys (+4dw pad/row)
    __shared__ __align__(16) unsigned short Plds[8][16][32];   // per-wave P tile

    const int tid  = threadIdx.x;
    const int lane = tid & 31;
    const int wave = tid >> 5;
    const int lm   = lane & 15;
    const int hi   = lane >> 4;
    const int kb   = hi * 8;

    const int qb0 = blockIdx.x * 128;
    const int bh  = blockIdx.y;
    const int q0w = qb0 + wave * 16;

    const size_t base_td = (size_t)bh * Tc * Dc;
    const size_t base_dt = (size_t)bh * Dc * Tc;
    const unsigned short* Kg = Kh + base_td;
    const unsigned short* Vg = Vt + base_dt;

    // Q fragments (two d-halves), loaded once
    FragBF qf0, qf1;
    {
        const unsigned short* qrow = Qh + base_td + (size_t)(q0w + lm) * Dc;
        qf0.q4[0] = *(const u32x4*)(qrow + kb);
        qf0.q4[1] = *(const u32x4*)(qrow + kb + 16);
        qf1.q4[0] = *(const u32x4*)(qrow + 32 + kb);
        qf1.q4[1] = *(const u32x4*)(qrow + 32 + kb + 16);
    }

    v8f o0 = {}, o1 = {}, o2 = {}, o3 = {};
    float mrun[8], lrun[8];
#pragma unroll
    for (int r = 0; r < 8; ++r) { mrun[r] = NEG_BIG; lrun[r] = 0.0f; }

    const int nch = (qb0 + 128) / 32;   // causal chunk count for this block

    // Prologue: TDM stage of chunk 0 into buffer 0 (wave 0 only; EXEC-independent DMA)
    if (wave == 0) {
        tdm_load_2d(lds_off(&Klds[0][0][0]), Kg, /*td0*/Dc, /*td1*/Tc,
                    /*tile*/Dc, 32, /*stride0*/Dc, /*pi: 32dw*/4, /*pa: 4dw*/3);
        tdm_load_2d(lds_off(&Vlds[0][0][0]), Vg, /*td0*/Tc, /*td1*/Dc,
                    /*tile*/32, Dc, /*stride0*/Tc, /*pi: 16dw*/3, /*pa: 4dw*/3);
    }

    for (int i = 0; i < nch; ++i) {
        const int cur = i & 1;
        const int k0  = i * 32;

        if (wave == 0) __builtin_amdgcn_s_wait_tensorcnt(0);
        __syncthreads();   // chunk i data visible to all waves

        // Kick off DMA for chunk i+1 into the other buffer (overlaps with compute)
        if (wave == 0 && (i + 1) < nch) {
            const int kn = k0 + 32;
            tdm_load_2d(lds_off(&Klds[cur ^ 1][0][0]), Kg + (size_t)kn * Dc,
                        Dc, Tc, Dc, 32, Dc, 4, 3);
            tdm_load_2d(lds_off(&Vlds[cur ^ 1][0][0]), Vg + kn,
                        Tc, Dc, 32, Dc, Tc, 3, 3);
        }

        if (k0 <= q0w + 15) {   // wave-uniform causal skip
            const unsigned short (*K)[72] = Klds[cur];
            const unsigned short (*V)[40] = Vlds[cur];

            // ---- S = Q K^T (left keys 0..15 / right keys 16..31) ----
            v8f sl = {}, sr = {};
            FragBF kf;
            kf.q4[0] = *(const u32x4*)&K[lm][kb];
            kf.q4[1] = *(const u32x4*)&K[lm][kb + 16];
            sl = wmma_bf16(qf0, kf, sl);
            kf.q4[0] = *(const u32x4*)&K[lm][32 + kb];
            kf.q4[1] = *(const u32x4*)&K[lm][32 + kb + 16];
            sl = wmma_bf16(qf1, kf, sl);
            kf.q4[0] = *(const u32x4*)&K[16 + lm][kb];
            kf.q4[1] = *(const u32x4*)&K[16 + lm][kb + 16];
            sr = wmma_bf16(qf0, kf, sr);
            kf.q4[0] = *(const u32x4*)&K[16 + lm][32 + kb];
            kf.q4[1] = *(const u32x4*)&K[16 + lm][32 + kb + 16];
            sr = wmma_bf16(qf1, kf, sr);

            // ---- online softmax ----
            float pl[8], pr[8];
#pragma unroll
            for (int r = 0; r < 8; ++r) {
                float s0 = sl[r] * 0.125f;
                float s1 = sr[r] * 0.125f;
                int q = q0w + hi * 8 + r;
                if (k0 + lm > q)      s0 = NEG_BIG;
                if (k0 + 16 + lm > q) s1 = NEG_BIG;
                float mx = fmaxf(s0, s1);
#pragma unroll
                for (int off = 1; off < 16; off <<= 1)
                    mx = fmaxf(mx, __shfl_xor(mx, off, 32));
                float nm = fmaxf(mrun[r], mx);
                float p0 = __expf(s0 - nm);
                float p1 = __expf(s1 - nm);
                float rs = p0 + p1;
#pragma unroll
                for (int off = 1; off < 16; off <<= 1)
                    rs += __shfl_xor(rs, off, 32);
                float alpha = __expf(mrun[r] - nm);
                lrun[r] = lrun[r] * alpha + rs;
                mrun[r] = nm;
                o0[r] *= alpha; o1[r] *= alpha; o2[r] *= alpha; o3[r] *= alpha;
                pl[r] = p0; pr[r] = p1;
            }

            // ---- P: C-layout -> A-layout via wave-private LDS ----
            unsigned short (*P)[32] = Plds[wave];
#pragma unroll
            for (int r = 0; r < 8; ++r) {
                int m = hi * 8 + r;
                P[m][lm]      = f2bf(pl[r]);
                P[m][16 + lm] = f2bf(pr[r]);
            }
            asm volatile("s_wait_dscnt 0x0" ::: "memory");   // LDS in-order per wave

            FragBF pf, vf;
            pf.q4[0] = *(const u32x4*)&P[lm][kb];
            pf.q4[1] = *(const u32x4*)&P[lm][kb + 16];
            vf.q4[0] = *(const u32x4*)&V[lm][kb];
            vf.q4[1] = *(const u32x4*)&V[lm][kb + 16];
            o0 = wmma_bf16(pf, vf, o0);
            vf.q4[0] = *(const u32x4*)&V[16 + lm][kb];
            vf.q4[1] = *(const u32x4*)&V[16 + lm][kb + 16];
            o1 = wmma_bf16(pf, vf, o1);
            vf.q4[0] = *(const u32x4*)&V[32 + lm][kb];
            vf.q4[1] = *(const u32x4*)&V[32 + lm][kb + 16];
            o2 = wmma_bf16(pf, vf, o2);
            vf.q4[0] = *(const u32x4*)&V[48 + lm][kb];
            vf.q4[1] = *(const u32x4*)&V[48 + lm][kb + 16];
            o3 = wmma_bf16(pf, vf, o3);
        }
        __syncthreads();   // all reads of buf cur done before it is refilled
    }

    // ---- epilogue: normalize and store to attn_out[B*T][E] (bf16) ----
    const int b = bh / Hc, h = bh % Hc;
#pragma unroll
    for (int r = 0; r < 8; ++r) {
        int q = q0w + hi * 8 + r;
        size_t row = (size_t)(b * Tc + q) * Ec + h * Dc;
        float inv = 1.0f / lrun[r];
        attn_out[row + lm]      = f2bf(o0[r] * inv);
        attn_out[row + 16 + lm] = f2bf(o1[r] * inv);
        attn_out[row + 32 + lm] = f2bf(o2[r] * inv);
        attn_out[row + 48 + lm] = f2bf(o3[r] * inv);
    }
}

// ---------------------------------------------------------------------------
// Kernel 3: output projection  out = attn_out @ w_proj + b_proj  (f32 out)
// ---------------------------------------------------------------------------
__global__ void proj_gemm(const unsigned short* __restrict__ a_bf,
                          const unsigned short* __restrict__ wt,
                          const float* __restrict__ bias,
                          float* __restrict__ out) {
    const int lane = threadIdx.x & 31;
    const int wave = threadIdx.x >> 5;
    const int m0 = blockIdx.x * 32 + (wave >> 2) * 16;
    const int n0 = blockIdx.y * 256 + (wave & 3) * 64;
    const int lm = lane & 15;
    const int hi = lane >> 4;
    const int kb = hi * 8;

    v8f acc0 = {}, acc1 = {}, acc2 = {}, acc3 = {};
    const unsigned short* arow = a_bf + (size_t)(m0 + lm) * Ec;

    for (int kk = 0; kk < Ec; kk += 32) {
        FragBF a;
        a.q4[0] = *(const u32x4*)(arow + kk + kb);
        a.q4[1] = *(const u32x4*)(arow + kk + kb + 16);
        FragBF b;
        const unsigned short* w0 = wt + (size_t)(n0 + lm) * Ec + kk;
        b.q4[0] = *(const u32x4*)(w0 + kb);
        b.q4[1] = *(const u32x4*)(w0 + kb + 16);
        acc0 = wmma_bf16(a, b, acc0);
        const unsigned short* w1 = wt + (size_t)(n0 + 16 + lm) * Ec + kk;
        b.q4[0] = *(const u32x4*)(w1 + kb);
        b.q4[1] = *(const u32x4*)(w1 + kb + 16);
        acc1 = wmma_bf16(a, b, acc1);
        const unsigned short* w2 = wt + (size_t)(n0 + 32 + lm) * Ec + kk;
        b.q4[0] = *(const u32x4*)(w2 + kb);
        b.q4[1] = *(const u32x4*)(w2 + kb + 16);
        acc2 = wmma_bf16(a, b, acc2);
        const unsigned short* w3 = wt + (size_t)(n0 + 48 + lm) * Ec + kk;
        b.q4[0] = *(const u32x4*)(w3 + kb);
        b.q4[1] = *(const u32x4*)(w3 + kb + 16);
        acc3 = wmma_bf16(a, b, acc3);
    }

#pragma unroll
    for (int nt = 0; nt < 4; ++nt) {
        v8f acc = (nt == 0) ? acc0 : (nt == 1) ? acc1 : (nt == 2) ? acc2 : acc3;
        int col = n0 + nt * 16 + lm;
        float bv = bias[col];
#pragma unroll
        for (int r = 0; r < 8; ++r) {
            int row = m0 + hi * 8 + r;
            out[(size_t)row * Ec + col] = acc[r] + bv;
        }
    }
}

// ---------------------------------------------------------------------------
// Launch
// ---------------------------------------------------------------------------
extern "C" void kernel_launch(void* const* d_in, const int* in_sizes, int n_in,
                              void* d_out, int out_size, void* d_ws, size_t ws_size,
                              hipStream_t stream) {
    const float* x      = (const float*)d_in[0];
    const float* w_attn = (const float*)d_in[1];
    const float* b_attn = (const float*)d_in[2];
    const float* w_proj = (const float*)d_in[3];
    const float* b_proj = (const float*)d_in[4];
    float* out = (float*)d_out;

    unsigned short* wt_attn = (unsigned short*)d_ws;
    unsigned short* wt_proj = wt_attn + (size_t)E3 * Ec;
    unsigned short* Qh      = wt_proj + (size_t)Ec * Ec;
    unsigned short* Kh      = Qh + (size_t)Mrows * Ec;
    unsigned short* Vt      = Kh + (size_t)Mrows * Ec;
    unsigned short* aout    = Vt + (size_t)Mrows * Ec;

    convert_weights<<<9216, 256, 0, stream>>>(w_attn, w_proj, wt_attn, wt_proj);
    qkv_gemm<<<dim3(Mrows / 32, E3 / 256), 256, 0, stream>>>(x, wt_attn, b_attn, Qh, Kh, Vt);
    attn_kernel<<<dim3(Tc / 128, Bc * Hc), 256, 0, stream>>>(Qh, Kh, Vt, aout);
    proj_gemm<<<dim3(Mrows / 32, Ec / 256), 256, 0, stream>>>(aout, wt_proj, b_proj, out);
}
